// Atten_RNN_4836133175827
// MI455X (gfx1250) — matrix operand order repaired and checked by reading, not compiled, not run on hardware
//
#include <hip/hip_runtime.h>
#include <math.h>

// ---------------------------------------------------------------------------
// MI455X (gfx1250) attention-RNN. All matrix math on v_wmma_f32_16x16x32_bf16.
// B=64, S=512, EMB=512, HID=1024, N_CLASS=32000.
// Round 2: software-pipelined (double-buffered) GEMM inner loops so
// global_load_b128 overlaps v_wmma instead of s_wait_loadcnt 0 every step.
// ---------------------------------------------------------------------------

typedef __attribute__((ext_vector_type(16))) __bf16        v16bf;
typedef __attribute__((ext_vector_type(8)))  float         v8f;
typedef __attribute__((ext_vector_type(4)))  unsigned int  u32x4;
typedef __attribute__((ext_vector_type(4)))  float         f32x4;

union ABf {                 // one WMMA A/B operand: 16 bf16 per lane
  v16bf        v;
  u32x4        q[2];
  unsigned int u[8];
};

__device__ __forceinline__ unsigned short f2bf(float f) {
  unsigned int u = __builtin_bit_cast(unsigned int, f);
  u += 0x7FFFu + ((u >> 16) & 1u);          // round-to-nearest-even
  return (unsigned short)(u >> 16);
}
__device__ __forceinline__ float bf2f(unsigned short s) {
  unsigned int u = ((unsigned int)s) << 16;
  return __builtin_bit_cast(float, u);
}
__device__ __forceinline__ unsigned int pack2(float a, float b) {
  return (unsigned int)f2bf(a) | ((unsigned int)f2bf(b) << 16);
}
__device__ __forceinline__ v8f wmma_bf16(const ABf& a, const ABf& b, v8f c) {
  return __builtin_amdgcn_wmma_f32_16x16x32_bf16(
      false, a.v, false, b.v, (short)0, c, false, false);
}
// A operand: lane holds row lane%16, contiguous K = base..base+15,
// base = (lane/16)*16 -> caller passes k already including base.
__device__ __forceinline__ void ldA(ABf& d, const unsigned short* row, int k) {
  d.q[0] = *(const u32x4*)(row + k);
  d.q[1] = *(const u32x4*)(row + k + 8);
}
// B operand: lane holds col lane%16, contiguous K = base..base+15,
// base = (lane/16)*16 -> caller passes k already including base.
__device__ __forceinline__ void ldB(ABf& d, const unsigned short* row, int k) {
  d.q[0] = *(const u32x4*)(row + k);
  d.q[1] = *(const u32x4*)(row + k + 8);
}

// --------------------------- prep kernels ----------------------------------

__global__ void k_cvt(const float* __restrict__ src, unsigned int* __restrict__ dst, int npairs) {
  for (int i = blockIdx.x * blockDim.x + threadIdx.x; i < npairs; i += gridDim.x * blockDim.x)
    dst[i] = pack2(src[2 * i], src[2 * i + 1]);
}

// gather emb[X[b,s]] -> Xemb_bf16[(s*64+b)*512 + e]
__global__ void k_gather(const int* __restrict__ X, const float* __restrict__ emb,
                         unsigned int* __restrict__ dst) {
  const int NP = 32768 * 256;                       // rows * (512/2) pairs
  for (int i = blockIdx.x * 256 + threadIdx.x; i < NP; i += gridDim.x * 256) {
    int m = i >> 8;                                 // row index = s*64+b
    int e = (i & 255) * 2;
    int s = m >> 6, b = m & 63;
    int t = X[b * 512 + s];
    const float* src = emb + (size_t)t * 512 + e;
    dst[i] = pack2(src[0], src[1]);
  }
}

__global__ void k_zero(unsigned int* __restrict__ p, int n) {
  int i = blockIdx.x * 256 + threadIdx.x;
  if (i < n) p[i] = 0u;
}

// ------------------- Xp = Xemb @ W_ih^T + b_ih + b_hh ----------------------
// M=32768, N=1024, K=512.  WG = 8 waves, wave owns 16(M) x 64(N).
__global__ void k_pregemm(const unsigned short* __restrict__ Xemb,
                          const unsigned short* __restrict__ Wih,
                          const float* __restrict__ b_ih, const float* __restrict__ b_hh,
                          unsigned short* __restrict__ Xp) {
  int w = threadIdx.x >> 5, lane = threadIdx.x & 31;
  int lr = lane & 15, lh = lane >> 4;
  int m0 = blockIdx.x * 128 + w * 16;
  int nb = blockIdx.y * 64;
  int ka = lh * 16, kb = lh * 16;
  const unsigned short* arow = Xemb + (size_t)(m0 + lr) * 512;
  const unsigned short* brow[4];
#pragma unroll
  for (int j = 0; j < 4; ++j) brow[j] = Wih + (size_t)(nb + j * 16 + lr) * 512;

  v8f z = {};
  v8f acc[4] = {z, z, z, z};
  ABf A0, A1, B0[4], B1[4];
  ldA(A0, arow, ka);
#pragma unroll
  for (int j = 0; j < 4; ++j) ldB(B0[j], brow[j], kb);

  for (int k = 0; k < 512; k += 64) {
    ldA(A1, arow, k + 32 + ka);
#pragma unroll
    for (int j = 0; j < 4; ++j) ldB(B1[j], brow[j], k + 32 + kb);
#pragma unroll
    for (int j = 0; j < 4; ++j) acc[j] = wmma_bf16(A0, B0[j], acc[j]);
    if (k + 64 < 512) {
      ldA(A0, arow, k + 64 + ka);
#pragma unroll
      for (int j = 0; j < 4; ++j) ldB(B0[j], brow[j], k + 64 + kb);
    }
#pragma unroll
    for (int j = 0; j < 4; ++j) acc[j] = wmma_bf16(A1, B1[j], acc[j]);
  }
#pragma unroll
  for (int j = 0; j < 4; ++j) {
    int n = nb + j * 16 + lr;
    float bias = b_ih[n] + b_hh[n];
#pragma unroll
    for (int r = 0; r < 8; ++r) {
      int m = m0 + r + lh * 8;
      Xp[(size_t)m * 1024 + n] = f2bf(acc[j][r] + bias);
    }
  }
}

// ---------------- one RNN step: Hcur = tanh(Xp_s + Hprev @ W_hh^T) ----------
// M=64, N=1024, K=1024. 16 WGs of 8 waves; wave owns 16(M) x 32(N).
__global__ void k_scan(const unsigned short* __restrict__ Hprev,
                       const unsigned short* __restrict__ Whh,
                       const unsigned short* __restrict__ Xps,
                       unsigned short* __restrict__ Hcur) {
  int w = threadIdx.x >> 5, lane = threadIdx.x & 31;
  int lr = lane & 15, lh = lane >> 4;
  int m0 = (w & 3) * 16;
  int n0 = blockIdx.x * 64 + (w >> 2) * 32;
  int ka = lh * 16, kb = lh * 16;
  const unsigned short* arow  = Hprev + (size_t)(m0 + lr) * 1024;
  const unsigned short* brow0 = Whh + (size_t)(n0 + lr) * 1024;
  const unsigned short* brow1 = Whh + (size_t)(n0 + 16 + lr) * 1024;

  v8f z = {};
  v8f acc0 = z, acc1 = z;
  ABf A0, B00, B10, A1, B01, B11;
  ldA(A0, arow, ka);
  ldB(B00, brow0, kb);
  ldB(B10, brow1, kb);

  for (int k = 0; k < 1024; k += 64) {
    ldA(A1, arow, k + 32 + ka);
    ldB(B01, brow0, k + 32 + kb);
    ldB(B11, brow1, k + 32 + kb);
    acc0 = wmma_bf16(A0, B00, acc0);
    acc1 = wmma_bf16(A0, B10, acc1);
    if (k + 64 < 1024) {
      ldA(A0, arow, k + 64 + ka);
      ldB(B00, brow0, k + 64 + kb);
      ldB(B10, brow1, k + 64 + kb);
    }
    acc0 = wmma_bf16(A1, B01, acc0);
    acc1 = wmma_bf16(A1, B11, acc1);
  }
#pragma unroll
  for (int r = 0; r < 8; ++r) {
    int m = m0 + r + lh * 8;
    int na = n0 + lr, nb2 = n0 + 16 + lr;
    float h0 = tanhf(acc0[r] + bf2f(Xps[(size_t)m * 1024 + na]));
    float h1 = tanhf(acc1[r] + bf2f(Xps[(size_t)m * 1024 + nb2]));
    Hcur[(size_t)m * 1024 + na]  = f2bf(h0);
    Hcur[(size_t)m * 1024 + nb2] = f2bf(h1);
  }
}

// ------------- scores[b,s] = <prev[s,b,:], last[b,:]> (wave per pair) -------
__global__ void k_scores(const unsigned short* __restrict__ Hb, float* __restrict__ scores) {
  int gw = blockIdx.x * 8 + (threadIdx.x >> 5);
  int lane = threadIdx.x & 31;
  if (gw >= 511 * 64) return;
  int s = gw / 64, b = gw % 64;
  const unsigned short* p = Hb + (size_t)(s + 1) * 65536 + (size_t)b * 1024;
  const unsigned short* q = Hb + (size_t)512 * 65536 + (size_t)b * 1024;
  float v = 0.0f;
  for (int i = lane; i < 1024; i += 32) v += bf2f(p[i]) * bf2f(q[i]);
  for (int off = 16; off; off >>= 1) v += __shfl_down(v, off, 32);
  if (lane == 0) scores[b * 511 + s] = v;
}

// --------------- softmax over s (len 511) per batch column ------------------
__global__ void k_softmax(float* __restrict__ sc) {
  __shared__ float red[256];
  int b = blockIdx.x, t = threadIdx.x;
  float* p = sc + b * 511;
  float e0 = p[t];
  float e1 = (t + 256 < 511) ? p[t + 256] : -3.4e38f;
  red[t] = fmaxf(e0, e1);
  __syncthreads();
  for (int s = 128; s; s >>= 1) { if (t < s) red[t] = fmaxf(red[t], red[t + s]); __syncthreads(); }
  float mx = red[0];
  __syncthreads();
  float x0 = expf(e0 - mx);
  float x1 = (t + 256 < 511) ? expf(e1 - mx) : 0.0f;
  red[t] = x0 + x1;
  __syncthreads();
  for (int s = 128; s; s >>= 1) { if (t < s) red[t] += red[t + s]; __syncthreads(); }
  float inv = 1.0f / red[0];
  p[t] = x0 * inv;
  if (t + 256 < 511) p[t + 256] = x1 * inv;
}

// ------- att_out[b,h] = sum_s attn * prev;  feat = [att_out | last] (bf16) --
__global__ void k_attout(const unsigned short* __restrict__ Hb, const float* __restrict__ attn,
                         unsigned short* __restrict__ feat) {
  int tid = blockIdx.x * 256 + threadIdx.x;        // 65536 = 64*1024
  int b = tid >> 10, h = tid & 1023;
  const float* a = attn + b * 511;
  const unsigned short* p = Hb + (size_t)65536 + (size_t)b * 1024 + h;
  float v0 = 0.0f, v1 = 0.0f, v2 = 0.0f, v3 = 0.0f;
  int s = 0;
  for (; s + 4 <= 511; s += 4) {
    v0 += a[s + 0] * bf2f(p[(size_t)(s + 0) * 65536]);
    v1 += a[s + 1] * bf2f(p[(size_t)(s + 1) * 65536]);
    v2 += a[s + 2] * bf2f(p[(size_t)(s + 2) * 65536]);
    v3 += a[s + 3] * bf2f(p[(size_t)(s + 3) * 65536]);
  }
  for (; s < 511; ++s) v0 += a[s] * bf2f(p[(size_t)s * 65536]);
  float v = (v0 + v1) + (v2 + v3);
  feat[(size_t)b * 2048 + h] = f2bf(v);
  feat[(size_t)b * 2048 + 1024 + h] = Hb[(size_t)512 * 65536 + (size_t)b * 1024 + h];
}

// ----------- out = feat @ W_out^T + b_out  (M=64, N=32000, K=2048) ----------
// Memory-bound on W_out (262MB fp32 read once @ 23.3TB/s): convert fp32->bf16
// in-register; double-buffer + prefetch to keep the HBM stream deep.
__global__ void k_out(const unsigned short* __restrict__ feat, const float* __restrict__ Wout,
                      const float* __restrict__ b_out, float* __restrict__ out) {
  int w = threadIdx.x >> 5, lane = threadIdx.x & 31;
  int lr = lane & 15, lh = lane >> 4;
  int m0 = (w & 3) * 16;
  int n0 = blockIdx.x * 32 + (w >> 2) * 16;
  int ka = lh * 16, kb = lh * 16;
  const unsigned short* arow = feat + (size_t)(m0 + lr) * 2048;
  const float* brow = Wout + (size_t)(n0 + lr) * 2048;

  v8f z = {};
  v8f acc = z;
  ABf A0, A1, Bm;
  f32x4 f0a, f1a, f2a, f3a, f0b, f1b, f2b, f3b;

  ldA(A0, arow, ka);
  f0a = *(const f32x4*)(brow + kb);
  f1a = *(const f32x4*)(brow + kb + 4);
  f2a = *(const f32x4*)(brow + kb + 8);
  f3a = *(const f32x4*)(brow + kb + 12);

  for (int k = 0; k < 2048; k += 64) {
    __builtin_prefetch(brow + k + 512, 0, 0);       // global_prefetch_b8
    ldA(A1, arow, k + 32 + ka);
    int kb1 = k + 32 + kb;
    f0b = *(const f32x4*)(brow + kb1);
    f1b = *(const f32x4*)(brow + kb1 + 4);
    f2b = *(const f32x4*)(brow + kb1 + 8);
    f3b = *(const f32x4*)(brow + kb1 + 12);

    Bm.u[0] = pack2(f0a.x, f0a.y); Bm.u[1] = pack2(f0a.z, f0a.w);
    Bm.u[2] = pack2(f1a.x, f1a.y); Bm.u[3] = pack2(f1a.z, f1a.w);
    Bm.u[4] = pack2(f2a.x, f2a.y); Bm.u[5] = pack2(f2a.z, f2a.w);
    Bm.u[6] = pack2(f3a.x, f3a.y); Bm.u[7] = pack2(f3a.z, f3a.w);
    acc = wmma_bf16(A0, Bm, acc);

    if (k + 64 < 2048) {
      ldA(A0, arow, k + 64 + ka);
      int kb2 = k + 64 + kb;
      f0a = *(const f32x4*)(brow + kb2);
      f1a = *(const f32x4*)(brow + kb2 + 4);
      f2a = *(const f32x4*)(brow + kb2 + 8);
      f3a = *(const f32x4*)(brow + kb2 + 12);
    }

    Bm.u[0] = pack2(f0b.x, f0b.y); Bm.u[1] = pack2(f0b.z, f0b.w);
    Bm.u[2] = pack2(f1b.x, f1b.y); Bm.u[3] = pack2(f1b.z, f1b.w);
    Bm.u[4] = pack2(f2b.x, f2b.y); Bm.u[5] = pack2(f2b.z, f2b.w);
    Bm.u[6] = pack2(f3b.x, f3b.y); Bm.u[7] = pack2(f3b.z, f3b.w);
    acc = wmma_bf16(A1, Bm, acc);
  }
  int n = n0 + lr;
  float bias = b_out[n];
#pragma unroll
  for (int r = 0; r < 8; ++r) {
    int m = m0 + r + lh * 8;
    out[(size_t)m * 32000 + n] = acc[r] + bias;
  }
}

// ------------------------------- launcher -----------------------------------

extern "C" void kernel_launch(void* const* d_in, const int* in_sizes, int n_in,
                              void* d_out, int out_size, void* d_ws, size_t ws_size,
                              hipStream_t stream) {
  const int*   X    = (const int*)d_in[0];
  const float* emb  = (const float*)d_in[1];
  const float* Wih  = (const float*)d_in[2];
  const float* Whh  = (const float*)d_in[3];
  const float* bih  = (const float*)d_in[4];
  const float* bhh  = (const float*)d_in[5];
  const float* Wout = (const float*)d_in[6];
  const float* bout = (const float*)d_in[7];
  float* out = (float*)d_out;

  char* ws = (char*)d_ws;
  // workspace layout (bytes)
  unsigned short* Wih_b = (unsigned short*)(ws + 0);           //  1,048,576
  unsigned short* Whh_b = (unsigned short*)(ws + 1048576);     //  2,097,152
  unsigned short* Xemb  = (unsigned short*)(ws + 3145728);     // 33,554,432
  unsigned short* Xp    = (unsigned short*)(ws + 36700160);    // 67,108,864
  unsigned short* Hb    = (unsigned short*)(ws + 103809024);   // 67,239,936 (513 steps)
  float*          sc    = (float*)(ws + 171048960);            //    130,816
  unsigned short* feat  = (unsigned short*)(ws + 171179776);   //    262,144

  // weight / embedding conversion to bf16, h_{-1} = 0
  k_cvt<<<256, 256, 0, stream>>>(Wih, (unsigned int*)Wih_b, 1024 * 512 / 2);
  k_cvt<<<512, 256, 0, stream>>>(Whh, (unsigned int*)Whh_b, 1024 * 1024 / 2);
  k_gather<<<4096, 256, 0, stream>>>(X, emb, (unsigned int*)Xemb);
  k_zero<<<128, 256, 0, stream>>>((unsigned int*)Hb, 32768);

  // hoisted input projection (parallel 34 GFLOP GEMM)
  k_pregemm<<<dim3(256, 16), 256, 0, stream>>>(Xemb, Wih_b, bih, bhh, Xp);

  // sequential recurrence: 512 small WMMA GEMM steps
  for (int s = 0; s < 512; ++s)
    k_scan<<<16, 256, 0, stream>>>(Hb + (size_t)s * 65536, Whh_b,
                                   Xp + (size_t)s * 65536,
                                   Hb + (size_t)(s + 1) * 65536);

  // attention + output
  k_scores<<<4088, 256, 0, stream>>>(Hb, sc);
  k_softmax<<<64, 256, 0, stream>>>(sc);
  k_attout<<<256, 256, 0, stream>>>(Hb, sc, feat);
  k_out<<<1000, 256, 0, stream>>>(feat, Wout, bout, out);
}
